// DownSample_8366596293055
// MI455X (gfx1250) — matrix operand order repaired and checked by reading, not compiled, not run on hardware
//
#include <hip/hip_runtime.h>

typedef __attribute__((ext_vector_type(16))) _Float16 v16h;
typedef __attribute__((ext_vector_type(8)))  float    v8f;

#define C_IN   32
#define C_OUT  64
#define K_TAPS 8
#define EPS    1e-5f

// d_ws layout (float units):
//   [0,32)    per-channel sum accumulator
//   [32,64)   per-channel sum-of-squares accumulator
//   [64,96)   scale  = gamma * rsqrt(var+eps)
//   [96,128)  bias   = beta - mean*scale
//   [128, 128+8192) packed f16 B fragments: [tap][ntile][lane][16] halfs
#define WS_SUM   0
#define WS_SQ    32
#define WS_SCALE 64
#define WS_BIAS  96
#define WS_WPACK 128
#define WPACK_HALFS (K_TAPS * 4 * 32 * 16)   // 16384 halfs = 32 KB

// ---------------------------------------------------------------- zero output
__global__ void zero_kernel(float* __restrict__ out, long long nElems,
                            float* __restrict__ ws) {
  long long stride = (long long)gridDim.x * blockDim.x;
  long long i = (long long)blockIdx.x * blockDim.x + threadIdx.x;
  long long n4 = nElems >> 2;
  float4* o4 = (float4*)out;
  float4 z; z.x = 0.f; z.y = 0.f; z.z = 0.f; z.w = 0.f;
  for (long long k = i; k < n4; k += stride) o4[k] = z;
  for (long long k = (n4 << 2) + i; k < nElems; k += stride) out[k] = 0.f;
  if (blockIdx.x == 0 && threadIdx.x < 64) ws[threadIdx.x] = 0.f;
}

// ------------------------------------------------- per-channel sum / sumsq
__global__ __launch_bounds__(256) void stats_kernel(
    const float* __restrict__ f, long long nElems, float* __restrict__ ws) {
  __shared__ float ssum[256];
  __shared__ float ssq[256];
  const int tid = threadIdx.x;
  long long i = (long long)blockIdx.x * 256 + tid;
  const long long stride = (long long)gridDim.x * 256;  // multiple of 32:
  float s = 0.f, q = 0.f;                               // channel fixed = tid&31
  for (; i < nElems; i += stride) {
    float v = f[i];
    s += v;
    q += v * v;
  }
  ssum[tid] = s;
  ssq[tid]  = q;
  __syncthreads();
  if (tid < 32) {
    float S = 0.f, Q = 0.f;
    for (int k = tid; k < 256; k += 32) { S += ssum[k]; Q += ssq[k]; }
    __hip_atomic_fetch_add(&ws[WS_SUM + tid], S, __ATOMIC_RELAXED,
                           __HIP_MEMORY_SCOPE_AGENT);
    __hip_atomic_fetch_add(&ws[WS_SQ + tid], Q, __ATOMIC_RELAXED,
                           __HIP_MEMORY_SCOPE_AGENT);
  }
}

// ------------------- finalize scale/bias + pack weights into WMMA B layout
// B layout (16-bit B 32x16, wave32): lane l holds column n = l&15,
// K = (l>>4)*8 + j for j < 8, 16 + (l>>4)*8 + (j-8) for j >= 8, packed in the lane's v16h.
__global__ void finalize_kernel(const float* __restrict__ gamma,
                                const float* __restrict__ beta,
                                const float* __restrict__ w, float invN,
                                float* __restrict__ ws) {
  const int tid = threadIdx.x;
  if (tid < 32) {
    float mean = ws[WS_SUM + tid] * invN;
    float var  = ws[WS_SQ + tid] * invN - mean * mean;
    float sc   = gamma[tid] * rsqrtf(var + EPS);
    ws[WS_SCALE + tid] = sc;
    ws[WS_BIAS + tid]  = beta[tid] - mean * sc;
  }
  _Float16* wp = (_Float16*)(ws + WS_WPACK);
  for (int i = tid; i < WPACK_HALFS; i += blockDim.x) {
    int j    = i & 15;
    int lane = (i >> 4) & 31;
    int nt   = (i >> 9) & 3;
    int t    = i >> 11;
    int hf   = lane >> 4;
    int n    = nt * 16 + (lane & 15);
    int k    = (j < 8) ? (hf * 8 + j) : (16 + hf * 8 + (j - 8));
    wp[i] = (_Float16)w[t * (C_IN * C_OUT) + k * C_OUT + n];
  }
}

// ------------------------------------------------------------- main kernel
// One wave = one tile of 16 points. A (16x32 f16) per ISA layout:
// lane l: row m = l&15; holds K in {hf*8..hf*8+7} then {16+hf*8..16+hf*8+7}
// where hf = l>>4. Masked per tap, 4 C tiles of 16x16 f32 accumulate C_OUT=64.
// B fragments are staged in LDS once per block so WMMA operands ride the
// ds pipe while the vmem pipe handles feature loads + scatter atomics.
__global__ __launch_bounds__(256) void conv_kernel(
    const float* __restrict__ feat, const int* __restrict__ out_index,
    const int* __restrict__ off_index, const float* __restrict__ ws,
    float* __restrict__ out, int nPoints) {
  __shared__ _Float16 bsh[WPACK_HALFS];  // 32 KB

  // ---- cooperative stage of packed B into LDS (before any wave may exit)
  {
    const float4* src = (const float4*)(ws + WS_WPACK);
    float4* dst = (float4*)bsh;
#pragma unroll
    for (int i = 0; i < (WPACK_HALFS / 8) / 256; ++i)
      dst[i * 256 + threadIdx.x] = src[i * 256 + threadIdx.x];
  }
  __syncthreads();

  const int lane = threadIdx.x & 31;
  const int wv   = threadIdx.x >> 5;
  const int tile = blockIdx.x * 8 + wv;
  const int p0   = tile * 16;
  if (p0 >= nPoints) return;  // wave-uniform: EXEC stays all-ones for WMMA

  const int m  = lane & 15;
  const int hf = lane >> 4;
  const int p  = p0 + m;
  const bool rowValid = (p < nPoints);
  const int  pc = rowValid ? p : (nPoints - 1);

  // ---- load + normalize + ReLU -> A fragment (f16)
  const float* fp = feat + (long long)pc * C_IN + hf * 8;
  float4 f0 = *(const float4*)(fp + 0);
  float4 f1 = *(const float4*)(fp + 4);
  float4 f2 = *(const float4*)(fp + 16);
  float4 f3 = *(const float4*)(fp + 20);
  const float* scb = ws + WS_SCALE + hf * 8;
  const float* bcb = ws + WS_BIAS + hf * 8;
  float4 s0 = *(const float4*)(scb + 0);
  float4 s1 = *(const float4*)(scb + 4);
  float4 s2 = *(const float4*)(scb + 16);
  float4 s3 = *(const float4*)(scb + 20);
  float4 b0 = *(const float4*)(bcb + 0);
  float4 b1 = *(const float4*)(bcb + 4);
  float4 b2 = *(const float4*)(bcb + 16);
  float4 b3 = *(const float4*)(bcb + 20);

  float fv[16] = {f0.x, f0.y, f0.z, f0.w, f1.x, f1.y, f1.z, f1.w,
                  f2.x, f2.y, f2.z, f2.w, f3.x, f3.y, f3.z, f3.w};
  float sv[16] = {s0.x, s0.y, s0.z, s0.w, s1.x, s1.y, s1.z, s1.w,
                  s2.x, s2.y, s2.z, s2.w, s3.x, s3.y, s3.z, s3.w};
  float bv[16] = {b0.x, b0.y, b0.z, b0.w, b1.x, b1.y, b1.z, b1.w,
                  b2.x, b2.y, b2.z, b2.w, b3.x, b3.y, b3.z, b3.w};

  v16h a;
#pragma unroll
  for (int j = 0; j < 16; ++j)
    a[j] = (_Float16)fmaxf(fmaf(fv[j], sv[j], bv[j]), 0.f);

  const int t_m = rowValid ? off_index[pc] : -1;  // -1 -> row contributes zero

  // ---- 8-tap masked WMMA accumulation, B from LDS
  const v16h* bfrag = (const v16h*)bsh;  // [tap*4 + ntile][lane]
  v8f c0 = {}, c1 = {}, c2 = {}, c3 = {};
  const v16h zeroA = {};
#pragma unroll
  for (int t = 0; t < K_TAPS; ++t) {
    v16h at = (t_m == t) ? a : zeroA;
    const v16h* bp = bfrag + (size_t)(t * 4) * 32 + lane;
    v16h bb0 = bp[0];
    v16h bb1 = bp[32];
    v16h bb2 = bp[64];
    v16h bb3 = bp[96];
    c0 = __builtin_amdgcn_wmma_f32_16x16x32_f16(false, at, false, bb0,
                                                (short)0, c0, false, false);
    c1 = __builtin_amdgcn_wmma_f32_16x16x32_f16(false, at, false, bb1,
                                                (short)0, c1, false, false);
    c2 = __builtin_amdgcn_wmma_f32_16x16x32_f16(false, at, false, bb2,
                                                (short)0, c2, false, false);
    c3 = __builtin_amdgcn_wmma_f32_16x16x32_f16(false, at, false, bb3,
                                                (short)0, c3, false, false);
  }

  // ---- scatter-add: C layout lane l: col = l&15, VGPR r -> row (l>>4)*8 + r
  const int rb = p0 + hf * 8;
  int idx[8];
  if (p0 + 16 <= nPoints) {
    int4 i0 = *(const int4*)(out_index + rb);
    int4 i1 = *(const int4*)(out_index + rb + 4);
    idx[0] = i0.x; idx[1] = i0.y; idx[2] = i0.z; idx[3] = i0.w;
    idx[4] = i1.x; idx[5] = i1.y; idx[6] = i1.z; idx[7] = i1.w;
  } else {
#pragma unroll
    for (int r = 0; r < 8; ++r)
      idx[r] = (rb + r < nPoints) ? out_index[rb + r] : 0;
  }
#pragma unroll
  for (int r = 0; r < 8; ++r) {
    if (rb + r >= nPoints) continue;
    float* ob = out + (long long)idx[r] * C_OUT + m;
    __hip_atomic_fetch_add(ob + 0,  c0[r], __ATOMIC_RELAXED, __HIP_MEMORY_SCOPE_AGENT);
    __hip_atomic_fetch_add(ob + 16, c1[r], __ATOMIC_RELAXED, __HIP_MEMORY_SCOPE_AGENT);
    __hip_atomic_fetch_add(ob + 32, c2[r], __ATOMIC_RELAXED, __HIP_MEMORY_SCOPE_AGENT);
    __hip_atomic_fetch_add(ob + 48, c3[r], __ATOMIC_RELAXED, __HIP_MEMORY_SCOPE_AGENT);
  }
}

// ---------------------------------------------------------------------------
extern "C" void kernel_launch(void* const* d_in, const int* in_sizes, int n_in,
                              void* d_out, int out_size, void* d_ws,
                              size_t ws_size, hipStream_t stream) {
  const float* features  = (const float*)d_in[0];
  const float* gamma     = (const float*)d_in[1];
  const float* beta      = (const float*)d_in[2];
  const float* weight    = (const float*)d_in[3];
  const int*   out_index = (const int*)d_in[4];
  const int*   off_index = (const int*)d_in[5];
  float* out = (float*)d_out;
  float* ws  = (float*)d_ws;

  const int nPoints = in_sizes[0] / C_IN;
  const long long nElemsIn = (long long)nPoints * C_IN;
  const long long nElemsOut = (long long)out_size;

  zero_kernel<<<2048, 256, 0, stream>>>(out, nElemsOut, ws);
  stats_kernel<<<2048, 256, 0, stream>>>(features, nElemsIn, ws);
  finalize_kernel<<<1, 256, 0, stream>>>(gamma, beta, weight,
                                         1.0f / (float)nPoints, ws);
  const int tiles  = (nPoints + 15) / 16;
  const int blocks = (tiles + 7) / 8;
  conv_kernel<<<blocks, 256, 0, stream>>>(features, out_index, off_index, ws,
                                          out, nPoints);
}
